// COIL_86148454023352
// MI455X (gfx1250) — matrix operand order repaired and checked by reading, not compiled, not run on hardware
//
#include <hip/hip_runtime.h>
#include <hip/hip_bf16.h>

// ---------------------------------------------------------------------------
// COIL scoring + softmax-CE loss for MI455X (gfx1250, wave32, WMMA).
// Shapes fixed by the reference: Bq=32 Lq=32 Bd=256 Ld=128 H=768 Dt=32 Dc=768.
// ---------------------------------------------------------------------------

typedef __attribute__((ext_vector_type(16))) __bf16 v16bf;
typedef __attribute__((ext_vector_type(8)))  __bf16 v8bf;
typedef __attribute__((ext_vector_type(8)))  float  v8f;

static __device__ __forceinline__ v8f wmma_bf16(v16bf a, v16bf b, v8f c) {
  // D = A(16x32 bf16) * B(32x16 bf16) + C(16x16 f32)
  return __builtin_amdgcn_wmma_f32_16x16x32_bf16(false, a, false, b, (short)0, c,
                                                 false, false);
}

static __device__ __forceinline__ v16bf cat8(v8bf lo, v8bf hi) {
  return __builtin_shufflevector(lo, hi, 0, 1, 2, 3, 4, 5, 6, 7,
                                         8, 9, 10, 11, 12, 13, 14, 15);
}

// Build an A-fragment (16x32, bf16) from a row-major fp32 matrix.
// Lane (hf = lane>>4, m = lane&15): elements 0..15 -> K = kb+16*hf+0..15
// (contiguous order, matching the B fragments).
static __device__ __forceinline__ v16bf load_a_f32(const float* ap) {
  v16bf a;
#pragma unroll
  for (int p = 0; p < 4; ++p) {
    float2 x = *(const float2*)(ap + 2 * p);
    float2 y = *(const float2*)(ap + 8 + 2 * p);
    a[2 * p]     = (__bf16)x.x;
    a[2 * p + 1] = (__bf16)x.y;
    a[8 + 2 * p]     = (__bf16)y.x;
    a[8 + 2 * p + 1] = (__bf16)y.y;
  }
  return a;
}

// A-fragment from a row-major bf16 matrix: two contiguous 16B runs per lane.
static __device__ __forceinline__ v16bf load_a_bf16(const __bf16* ap) {
  v8bf lo = *(const v8bf*)ap;
  v8bf hi = *(const v8bf*)(ap + 8);
  return cat8(lo, hi);
}

// ---------------------------------------------------------------------------
// 0) Pack weight W[K][N] (fp32, row-major) into B-fragment layout:
//    PW[k/16][n][k%16] (bf16) so each lane's B load is one contiguous 32B.
// ---------------------------------------------------------------------------
__global__ void pack_w_kernel(const float* __restrict__ W,
                              __bf16* __restrict__ PW, int K, int N) {
  int tid = blockIdx.x * blockDim.x + threadIdx.x;
  if (tid >= K * N) return;
  int k = tid / N, n = tid - k * N;
  PW[(size_t)(k >> 4) * (N * 16) + (size_t)n * 16 + (k & 15)] =
      (__bf16)W[(size_t)k * N + n];
}

// ---------------------------------------------------------------------------
// 1) Token projection: out[M][32] = relu(hid[M][768] @ W_tok + b_tok), bf16.
//    4 waves/block, each wave owns a 16-row tile, 2 col-tiles (N=32), K=768.
// ---------------------------------------------------------------------------
__global__ void proj_tok_kernel(const float* __restrict__ hid,
                                const __bf16* __restrict__ PW,  // [48][32][16]
                                const float* __restrict__ bias,
                                __bf16* __restrict__ out) {
  const int lane = threadIdx.x & 31;
  const int wave = threadIdx.x >> 5;
  const int rowbase = (blockIdx.x * 4 + wave) * 16;
  const int hf = lane >> 4, m = lane & 15;

  v8f c0 = {}, c1 = {};
  const float* arow = hid + (size_t)(rowbase + m) * 768;
#pragma unroll 4
  for (int kb = 0; kb < 768; kb += 32) {
    v16bf a = load_a_f32(arow + kb + 16 * hf);
    int kblk = (kb >> 4) + hf;
    v16bf b0 = *(const v16bf*)(PW + ((size_t)kblk * 32 + m) * 16);
    v16bf b1 = *(const v16bf*)(PW + ((size_t)kblk * 32 + 16 + m) * 16);
    c0 = wmma_bf16(a, b0, c0);
    c1 = wmma_bf16(a, b1, c1);
  }
  float bias0 = bias[m], bias1 = bias[16 + m];
#pragma unroll
  for (int r = 0; r < 8; ++r) {
    int row = rowbase + r + 8 * hf;
    float v0 = fmaxf(c0[r] + bias0, 0.0f);
    float v1 = fmaxf(c1[r] + bias1, 0.0f);
    out[(size_t)row * 32 + m]      = (__bf16)v0;
    out[(size_t)row * 32 + 16 + m] = (__bf16)v1;
  }
}

// ---------------------------------------------------------------------------
// 2) CLS projection: out[M][768] = hid[:,0,:] @ W_cls + b_cls (bf16 out).
//    One wave per 16x16 output tile; row_stride selects token-0 rows.
// ---------------------------------------------------------------------------
__global__ void proj_cls_kernel(const float* __restrict__ hid0, int row_stride,
                                const __bf16* __restrict__ PW,  // [48][768][16]
                                const float* __restrict__ bias,
                                __bf16* __restrict__ out) {
  const int lane = threadIdx.x & 31;
  const int hf = lane >> 4, m = lane & 15;
  const int rowbase = blockIdx.x * 16;
  const int nbase = blockIdx.y * 16;

  v8f c = {};
  const float* arow = hid0 + (size_t)(rowbase + m) * row_stride;
#pragma unroll 4
  for (int kb = 0; kb < 768; kb += 32) {
    v16bf a = load_a_f32(arow + kb + 16 * hf);
    int kblk = (kb >> 4) + hf;
    v16bf b = *(const v16bf*)(PW + ((size_t)kblk * 768 + nbase + m) * 16);
    c = wmma_bf16(a, b, c);
  }
  float bv = bias[nbase + m];
#pragma unroll
  for (int r = 0; r < 8; ++r)
    out[(size_t)(rowbase + r + 8 * hf) * 768 + nbase + m] = (__bf16)(c[r] + bv);
}

// ---------------------------------------------------------------------------
// 3) cls_scores[32][256] = qcls @ dcls^T (K=768).  B = dcls^T means each
//    lane's B strip is a contiguous run of a dcls row.
// ---------------------------------------------------------------------------
__global__ void cls_score_kernel(const __bf16* __restrict__ q,   // [32][768]
                                 const __bf16* __restrict__ d,   // [256][768]
                                 float* __restrict__ out) {      // [32][256]
  const int lane = threadIdx.x & 31;
  const int hf = lane >> 4, m = lane & 15;
  const int rowbase = blockIdx.x * 16;
  const int nbase = blockIdx.y * 16;

  v8f c = {};
  const __bf16* arow = q + (size_t)(rowbase + m) * 768;
  const __bf16* brow = d + (size_t)(nbase + m) * 768;
#pragma unroll 4
  for (int kb = 0; kb < 768; kb += 32) {
    v16bf a = load_a_bf16(arow + kb + 16 * hf);
    v16bf b = *(const v16bf*)(brow + kb + 16 * hf);
    c = wmma_bf16(a, b, c);
  }
#pragma unroll
  for (int r = 0; r < 8; ++r)
    out[(size_t)(rowbase + r + 8 * hf) * 256 + nbase + m] = c[r];
}

// ---------------------------------------------------------------------------
// 4) Token interaction. One wave per (query a, doc b):
//    16 WMMAs (2 q-row tiles x 8 doc-col tiles, K=32 in ONE wmma each),
//    exact-match masked max over doc tokens, qmask-weighted sum over query
//    tokens, plus cls score -> final scores[a][b].
// ---------------------------------------------------------------------------
__global__ void interact_kernel(const __bf16* __restrict__ qreps,  // [1024][32]
                                const __bf16* __restrict__ dreps,  // [32768][32]
                                const int* __restrict__ qids,      // [32][32]
                                const int* __restrict__ dids,      // [256][128]
                                const int* __restrict__ qam,       // [32][32]
                                const float* __restrict__ clss,    // [32][256]
                                float* __restrict__ scores) {      // [32][256]
  const int a = blockIdx.x, b = blockIdx.y;
  const int lane = threadIdx.x & 31;
  const int hf = lane >> 4, m = lane & 15;
  const float NEG_INF = -__builtin_huge_valf();

  // last attended position for this query (mask_sep, no_sep=True)
  int sep;
  {
    int s = 0;
    for (int i = 0; i < 32; ++i) s += qam[a * 32 + i];
    sep = s - 1;
  }

  // A fragments: q-token rows [a*32 .. a*32+15] and [a*32+16 .. a*32+31]
  v16bf A0 = load_a_bf16(qreps + (size_t)(a * 32 + m) * 32 + 16 * hf);
  v16bf A1 = load_a_bf16(qreps + (size_t)(a * 32 + 16 + m) * 32 + 16 * hf);

  int qid0[8], qid1[8];
#pragma unroll
  for (int r = 0; r < 8; ++r) {
    qid0[r] = qids[a * 32 + 8 * hf + r];
    qid1[r] = qids[a * 32 + 16 + 8 * hf + r];
  }

  float m0[8], m1[8];
#pragma unroll
  for (int r = 0; r < 8; ++r) { m0[r] = NEG_INF; m1[r] = NEG_INF; }

  for (int jt = 0; jt < 8; ++jt) {
    // B fragment: doc-token column j = jt*16 + m, K strip contiguous.
    v16bf B = *(const v16bf*)(dreps + (size_t)(b * 128 + jt * 16 + m) * 32 +
                              16 * hf);
    int did = dids[b * 128 + jt * 16 + m];
    v8f z0 = {}, z1 = {};
    v8f c0 = wmma_bf16(A0, B, z0);
    v8f c1 = wmma_bf16(A1, B, z1);
#pragma unroll
    for (int r = 0; r < 8; ++r) {
      // max_j (score * exact_match): mismatched entries contribute 0.
      m0[r] = fmaxf(m0[r], (qid0[r] == did) ? c0[r] : 0.0f);
      m1[r] = fmaxf(m1[r], (qid1[r] == did) ? c1[r] : 0.0f);
    }
  }

  // Max over the 16 doc-token columns spread across lanes of each half.
#pragma unroll
  for (int off = 1; off < 16; off <<= 1) {
#pragma unroll
    for (int r = 0; r < 8; ++r) {
      m0[r] = fmaxf(m0[r], __shfl_xor(m0[r], off, 16));
      m1[r] = fmaxf(m1[r], __shfl_xor(m1[r], off, 16));
    }
  }

  // qmask-weighted sum over query tokens (drop i==0 and i==sep).
  float part = 0.0f;
#pragma unroll
  for (int r = 0; r < 8; ++r) {
    int i0 = r + 8 * hf;
    int i1 = 16 + r + 8 * hf;
    float w0 = (i0 > 0 && i0 != sep) ? (float)qam[a * 32 + i0] : 0.0f;
    float w1 = (i1 > 0 && i1 != sep) ? (float)qam[a * 32 + i1] : 0.0f;
    part += m0[r] * w0 + m1[r] * w1;
  }
  part += __shfl_xor(part, 16, 32);  // combine the two lane halves

  if (lane == 0) scores[a * 256 + b] = part + clss[a * 256 + b];
}

// ---------------------------------------------------------------------------
// 5) Softmax cross-entropy loss, labels = arange(32)*8. One wave, lane=row.
// ---------------------------------------------------------------------------
__global__ void loss_kernel(const float* __restrict__ scores,  // [32][256]
                            float* __restrict__ loss) {
  const int a = threadIdx.x;  // 0..31
  const float* row = scores + a * 256;
  float mx = -__builtin_huge_valf();
  for (int b = 0; b < 256; ++b) mx = fmaxf(mx, row[b]);
  float se = 0.0f;
  for (int b = 0; b < 256; ++b) se += expf(row[b] - mx);
  float lp = row[a * 8] - mx - logf(se);
#pragma unroll
  for (int off = 1; off < 32; off <<= 1) lp += __shfl_xor(lp, off, 32);
  if (a == 0) loss[0] = -lp * (1.0f / 32.0f);
}

// ---------------------------------------------------------------------------
extern "C" void kernel_launch(void* const* d_in, const int* in_sizes, int n_in,
                              void* d_out, int out_size, void* d_ws,
                              size_t ws_size, hipStream_t stream) {
  (void)in_sizes; (void)n_in; (void)out_size; (void)ws_size;

  const float* qry_hidden = (const float*)d_in[0];  // [32][32][768]
  const float* doc_hidden = (const float*)d_in[1];  // [256][128][768]
  const float* W_tok      = (const float*)d_in[2];  // [768][32]
  const float* b_tok      = (const float*)d_in[3];  // [32]
  const float* W_cls      = (const float*)d_in[4];  // [768][768]
  const float* b_cls      = (const float*)d_in[5];  // [768]
  const int*   qids       = (const int*)d_in[6];    // [32][32]
  const int*   dids       = (const int*)d_in[7];    // [256][128]
  const int*   qam        = (const int*)d_in[8];    // [32][32]
  float* out = (float*)d_out;                       // [1 + 32*256]

  char* ws = (char*)d_ws;  // all offsets 256B-aligned
  __bf16* PW_tok = (__bf16*)(ws + 0);        //   49152 B
  __bf16* PW_cls = (__bf16*)(ws + 49152);    // 1179648 B
  __bf16* qreps  = (__bf16*)(ws + 1228800);  //   65536 B
  __bf16* dreps  = (__bf16*)(ws + 1294336);  // 2097152 B
  __bf16* qcls   = (__bf16*)(ws + 3391488);  //   49152 B
  __bf16* dcls   = (__bf16*)(ws + 3440640);  //  393216 B
  float*  clss   = (float*)(ws + 3833856);   //   32768 B

  pack_w_kernel<<<(768 * 32 + 255) / 256, 256, 0, stream>>>(W_tok, PW_tok, 768, 32);
  pack_w_kernel<<<(768 * 768 + 255) / 256, 256, 0, stream>>>(W_cls, PW_cls, 768, 768);

  proj_tok_kernel<<<1024 / 64, 128, 0, stream>>>(qry_hidden, PW_tok, b_tok, qreps);
  proj_tok_kernel<<<32768 / 64, 128, 0, stream>>>(doc_hidden, PW_tok, b_tok, dreps);

  proj_cls_kernel<<<dim3(2, 48), 32, 0, stream>>>(qry_hidden, 32 * 768, PW_cls,
                                                  b_cls, qcls);
  proj_cls_kernel<<<dim3(16, 48), 32, 0, stream>>>(doc_hidden, 128 * 768, PW_cls,
                                                   b_cls, dcls);

  cls_score_kernel<<<dim3(2, 16), 32, 0, stream>>>(qcls, dcls, clss);

  interact_kernel<<<dim3(32, 256), 32, 0, stream>>>(qreps, dreps, qids, dids,
                                                    qam, clss, out + 1);

  loss_kernel<<<1, 32, 0, stream>>>(out + 1, out);
}
